// STMetaAGCRU_69054484185220
// MI455X (gfx1250) — matrix operand order repaired and checked by reading, not compiled, not run on hardware
//
#include <hip/hip_runtime.h>
#include <hip/hip_bf16.h>

// ---------------------------------------------------------------------------
// STMetaAGCRU on MI455X (gfx1250): bf16 WMMA GEMMs with fused GRU epilogues.
// v_wmma_f32_16x16x32_bf16 (wave32), double-buffered LDS-staged B fragments,
// 2 row-tiles per wave, pre-swizzled bf16 A fragments when workspace allows.
// ---------------------------------------------------------------------------

typedef __attribute__((ext_vector_type(16))) __bf16 v16bf;
typedef __attribute__((ext_vector_type(8)))  float  v8f;

static constexpr int Bb   = 16;    // batch
static constexpr int Tt   = 12;    // timesteps
static constexpr int Nn   = 1024;  // graph nodes
static constexpr int Hh   = 64;    // hidden
static constexpr int Kk   = 3;     // Chebyshev order
static constexpr int DIN  = 65;    // C + H
static constexpr int KT   = 32;    // 1024/32 K-tiles
static constexpr int NT_G    = 4;  // 64/16  (h-only columns, x handled separately)
static constexpr int NT_GATE = 8;  // 128/16
static constexpr int NT_UPD  = 4;  // 64/16

// -------------------------- workspace layout (bytes) ------------------------
static constexpr size_t OFF_H   = 0;                         // 4 MB fp32 h
static constexpr size_t OFF_Z   = OFF_H  + (size_t)Bb*Nn*Hh*4;
static constexpr size_t OFF_R   = OFF_Z  + (size_t)Bb*Nn*Hh*4;
static constexpr size_t OFF_HB  = OFF_R  + (size_t)Bb*Nn*Hh*4;          // 2 MB bf16 frags
static constexpr size_t OFF_VF  = OFF_HB + (size_t)Bb*KT*NT_G*512*2;    // 4 MB bf16 frags
static constexpr size_t OFF_XS  = OFF_VF + (size_t)Bb*KT*NT_GATE*512*2; // 3 MB fp32
static constexpr size_t OFF_XF  = OFF_XS + (size_t)Bb*Kk*Nn*16*4;       // 0.5 MB bf16 frags
static constexpr size_t OFF_S1  = OFF_XF + (size_t)Bb*KT*512*2;         // 12 MB fp32
static constexpr size_t OFF_GF  = OFF_S1 + (size_t)Bb*Kk*Nn*Hh*4;       // 6 MB bf16 Gfrag
static constexpr size_t OFF_AF  = OFF_GF + (size_t)Kk*Nn*Nn*2;          // 32 MB bf16 Afrag
static constexpr size_t WS_FULL = OFF_AF + (size_t)Bb*Nn*Nn*2;

__device__ __forceinline__ unsigned short f2bf(float f) {      // RNE
  unsigned int u = __float_as_uint(f);
  u = u + 0x7FFFu + ((u >> 16) & 1u);
  return (unsigned short)(u >> 16);
}
// pack bf16(lo),bf16(hi) by truncation: one v_perm_b32
__device__ __forceinline__ unsigned int packbf_tr(float lo, float hi) {
  return __builtin_amdgcn_perm(__float_as_uint(hi), __float_as_uint(lo), 0x07060302u);
}

union BF16Frag { v16bf v; unsigned short u[16]; unsigned int w[8]; };

// A-fragment (16x32 bf16) from row-major fp32 (fallback path, v_perm packing).
// ISA layout: lanes 0-15 (M=lane): K={0..7,16..23}; lanes 16-31: K={8..15,24..31}.
__device__ __forceinline__ v16bf load_afrag_f32(const float* __restrict__ Abase,
                                                int row0, int k0, int lda, int lane) {
  const int g = lane >> 4, m = lane & 15;
  const float* p = Abase + (size_t)(row0 + m) * lda + k0 + g * 8;
  float4 f0 = *(const float4*)(p);
  float4 f1 = *(const float4*)(p + 4);
  float4 f2 = *(const float4*)(p + 16);
  float4 f3 = *(const float4*)(p + 20);
  BF16Frag a;
  a.w[0] = packbf_tr(f0.x, f0.y); a.w[1] = packbf_tr(f0.z, f0.w);
  a.w[2] = packbf_tr(f1.x, f1.y); a.w[3] = packbf_tr(f1.z, f1.w);
  a.w[4] = packbf_tr(f2.x, f2.y); a.w[5] = packbf_tr(f2.z, f2.w);
  a.w[6] = packbf_tr(f3.x, f3.y); a.w[7] = packbf_tr(f3.z, f3.w);
  return a.v;
}

// B-fragment element position: lanes 0-15 hold K={0..7,16..23}, lanes 16-31 K={8..15,24..31}.
__device__ __forceinline__ int bfrag_pos(int kk, int n) {
  return (n + (((kk >> 3) & 1) << 4)) * 16 + (kk & 7) + ((kk >> 4) << 3);
}

// ---------------------------------------------------------------------------
// Setup / elementwise kernels
// ---------------------------------------------------------------------------

__global__ void zero_f32_kernel(float* __restrict__ p, int n) {
  int i = blockIdx.x * blockDim.x + threadIdx.x;
  if (i < n) p[i] = 0.f;
}

// fp32 matrices -> bf16 A-fragment layout (RNE), nmat matrices of 1024x1024.
__global__ void conv_afrag_kernel(const float* __restrict__ src,
                                  unsigned short* __restrict__ dst, int total) {
  int i = blockIdx.x * blockDim.x + threadIdx.x;
  if (i >= total) return;
  int mat = i >> 20, idx = i & ((1 << 20) - 1);
  int row = idx >> 10, col = idx & 1023;
  int rt = row >> 4, m = row & 15, kt = col >> 5, kk = col & 31;
  int g = (kk >> 3) & 1;
  int e = (kk & 7) + ((kk & 16) ? 8 : 0);
  dst[((size_t)mat << 20) + (size_t)(rt * KT + kt) * 512 + (m + (g << 4)) * 16 + e] =
      f2bf(src[i]);
}

// x (B,T,N) -> bf16 B-fragments, K=node, N=timestep padded 12->16.
__global__ void build_xfrag_kernel(const float* __restrict__ x,
                                   unsigned short* __restrict__ xF) {
  int i = blockIdx.x * blockDim.x + threadIdx.x;
  if (i >= Bb * Nn * 16) return;
  int tc = i & 15, j = (i >> 4) & (Nn - 1), b = i >> 14;
  float val = (tc < Tt) ? x[((size_t)(b * Tt + tc) << 10) + j] : 0.f;
  int kt = j >> 5, kk = j & 31;
  xF[(size_t)b * (KT * 512) + (size_t)kt * 512 + bfrag_pos(kk, tc)] = f2bf(val);
}

// h (gate) or r*h (update) -> bf16 B-fragments (K=node, N=hidden, 4 tiles).
__global__ void build_hB_kernel(const float* __restrict__ h, const float* __restrict__ r,
                                unsigned short* __restrict__ hB, int use_r) {
  int i = blockIdx.x * blockDim.x + threadIdx.x;
  if (i >= Bb * Nn * Hh) return;
  int c = i & 63, j = (i >> 6) & (Nn - 1), b = i >> 16;
  float val = use_r ? (r[i] * h[i]) : h[i];
  int kt = j >> 5, kk = j & 31, nt = c >> 4, n = c & 15;
  hB[(size_t)b * (KT * NT_G * 512) + (size_t)(kt * NT_G + nt) * 512 + bfrag_pos(kk, n)] =
      f2bf(val);
}

// V = S1_flat @ W (K=195), x-term from precomputed XS; output as bf16 B-frags.
__global__ void small_gemm_kernel(const float* __restrict__ S1h, const float* __restrict__ XS,
                                  const float* __restrict__ W, unsigned short* __restrict__ vF,
                                  int nout, int t) {
  int i = blockIdx.x * blockDim.x + threadIdx.x;
  if (i >= Bb * Nn * nout) return;
  int po  = i % nout;
  int row = (i / nout) & (Nn - 1);
  int b   = i / (nout * Nn);
  float acc = 0.f;
  for (int k = 0; k < Kk; ++k) {
    const float* wrow = W + (size_t)(k * DIN) * nout + po;
    size_t base = ((size_t)(b * Kk + k) << 10) + row;
    acc += XS[base * 16 + t] * wrow[0];                 // x column (p = 0)
    const float* sp = S1h + base * Hh;
    for (int p = 0; p < Hh; ++p)                        // h columns (p = 1..64)
      acc += sp[p] * wrow[(size_t)(p + 1) * nout];
  }
  int ntiles = nout >> 4;
  int kt = row >> 5, kk = row & 31, nt = po >> 4, n = po & 15;
  vF[(size_t)b * (KT * ntiles * 512) + (size_t)(kt * ntiles + nt) * 512 + bfrag_pos(kk, n)] =
      f2bf(acc);
}

// out = h @ W_dec + b_dec, transposed to (B, 12, N, 1)
__global__ void decode_kernel(const float* __restrict__ h, const float* __restrict__ Wd,
                              const float* __restrict__ bd, float* __restrict__ out) {
  int i = blockIdx.x * blockDim.x + threadIdx.x;
  if (i >= Bb * Nn) return;
  int n = i & (Nn - 1), b = i >> 10;
  const float* hr = h + (size_t)i * Hh;
  for (int s = 0; s < Tt; ++s) {
    float acc = bd[s];
    for (int hh = 0; hh < Hh; ++hh) acc += hr[hh] * Wd[hh * Tt + s];
    out[((size_t)(b * Tt + s) << 10) + n] = acc;
  }
}

// ---------------------------------------------------------------------------
// WMMA GEMM core: 128 threads = 4 waves; each wave owns TWO 16-row strips
// (rt and rt+32) across all column tiles.  B tiles staged through a
// double-buffered LDS pipeline: one barrier per K-step, global loads for
// tile kt+1 overlap the WMMAs of tile kt.
// ---------------------------------------------------------------------------

template <int NT, bool FRAG>
__device__ __forceinline__ void gemm_body2(const void* __restrict__ Ab,
                                           const unsigned short* __restrict__ Bbase,
                                           unsigned short* sB,   // 2 * NT * 512 ushorts
                                           int rt0, int rt1, int lane,
                                           v8f* acc0, v8f* acc1) {
  const int tid = threadIdx.x;
  {  // preload K-tile 0 into buffer 0
    const unsigned int* gsrc = (const unsigned int*)Bbase;
    unsigned int* sdst = (unsigned int*)sB;
#pragma unroll
    for (int c = 0; c < NT * 2; ++c) sdst[c * 128 + tid] = gsrc[c * 128 + tid];
  }
  __syncthreads();
  unsigned int stage[NT * 2];
  for (int kt = 0; kt < KT; ++kt) {
    const int cur = (kt & 1) * (NT * 512);
    const int nxt = ((kt + 1) & 1) * (NT * 512);
    if (kt + 1 < KT) {  // fetch next B tile into registers (overlaps WMMA below)
      const unsigned int* gsrc = (const unsigned int*)(Bbase + (size_t)(kt + 1) * NT * 512);
#pragma unroll
      for (int c = 0; c < NT * 2; ++c) stage[c] = gsrc[c * 128 + tid];
    }
    v16bf a0, a1;
    if (FRAG) {
      const unsigned short* af = (const unsigned short*)Ab;
      a0 = *(const v16bf*)(af + (size_t)(rt0 * KT + kt) * 512 + lane * 16);
      a1 = *(const v16bf*)(af + (size_t)(rt1 * KT + kt) * 512 + lane * 16);
    } else {
      a0 = load_afrag_f32((const float*)Ab, rt0 * 16, kt * 32, Nn, lane);
      a1 = load_afrag_f32((const float*)Ab, rt1 * 16, kt * 32, Nn, lane);
    }
#pragma unroll
    for (int nt = 0; nt < NT; ++nt) {
      v16bf bf = *(const v16bf*)(sB + cur + nt * 512 + lane * 16);
      acc0[nt] = __builtin_amdgcn_wmma_f32_16x16x32_bf16(false, a0, false, bf,
                                                         (short)0, acc0[nt], false, false);
      acc1[nt] = __builtin_amdgcn_wmma_f32_16x16x32_bf16(false, a1, false, bf,
                                                         (short)0, acc1[nt], false, false);
    }
    if (kt + 1 < KT) {  // commit next tile to the alternate buffer
      unsigned int* sdst = (unsigned int*)(sB + nxt);
#pragma unroll
      for (int c = 0; c < NT * 2; ++c) sdst[c * 128 + tid] = stage[c];
    }
    __syncthreads();
  }
}

// S1h[b,k] = G[k] @ Hmat[b]   (M=1024, K=1024, N=64)
template <bool FRAG>
__global__ __launch_bounds__(128) void gemm_G_kernel(const void* __restrict__ G,
                                                     const unsigned short* __restrict__ hB,
                                                     float* __restrict__ S1h) {
  __shared__ __align__(32) unsigned short sB[2 * NT_G * 512];
  const int lane = threadIdx.x & 31, w = threadIdx.x >> 5;
  const int rt0 = blockIdx.x * 4 + w, rt1 = rt0 + 32;
  const int k = blockIdx.y, b = blockIdx.z;
  const void* Ab = FRAG ? (const void*)((const unsigned short*)G + ((size_t)k << 20))
                        : (const void*)((const float*)G + ((size_t)k << 20));
  const unsigned short* Bbase = hB + (size_t)b * (KT * NT_G * 512);
  const v8f vz = {0.f, 0.f, 0.f, 0.f, 0.f, 0.f, 0.f, 0.f};
  v8f acc0[NT_G], acc1[NT_G];
#pragma unroll
  for (int nt = 0; nt < NT_G; ++nt) { acc0[nt] = vz; acc1[nt] = vz; }
  gemm_body2<NT_G, FRAG>(Ab, Bbase, sB, rt0, rt1, lane, acc0, acc1);
  const int g = lane >> 4, n = lane & 15;
#pragma unroll
  for (int mw = 0; mw < 2; ++mw) {
    const int rbase = (mw ? rt1 : rt0) * 16;
    const v8f* acc = mw ? acc1 : acc0;
#pragma unroll
    for (int nt = 0; nt < NT_G; ++nt)
#pragma unroll
      for (int rr = 0; rr < 8; ++rr) {
        int row = rbase + g * 8 + rr;
        S1h[(((size_t)(b * Kk + k) << 10) + row) * Hh + nt * 16 + n] = acc[nt][rr];
      }
  }
}

// XS[b,k,:,t] = G[k] @ Xcols[b]   (N=16, once per call; fp32 A path)
__global__ __launch_bounds__(128) void gemm_X_kernel(const float* __restrict__ G,
                                                     const unsigned short* __restrict__ xF,
                                                     float* __restrict__ XS) {
  __shared__ __align__(32) unsigned short sB[2 * 512];
  const int lane = threadIdx.x & 31, w = threadIdx.x >> 5;
  const int rt0 = blockIdx.x * 4 + w, rt1 = rt0 + 32;
  const int k = blockIdx.y, b = blockIdx.z;
  const float* Ab = G + ((size_t)k << 20);
  const unsigned short* Bbase = xF + (size_t)b * (KT * 512);
  const v8f vz = {0.f, 0.f, 0.f, 0.f, 0.f, 0.f, 0.f, 0.f};
  v8f acc0 = vz, acc1 = vz;
  gemm_body2<1, false>((const void*)Ab, Bbase, sB, rt0, rt1, lane, &acc0, &acc1);
  const int g = lane >> 4, n = lane & 15;
#pragma unroll
  for (int mw = 0; mw < 2; ++mw) {
    const int rbase = (mw ? rt1 : rt0) * 16;
    const v8f* acc = mw ? &acc1 : &acc0;
#pragma unroll
    for (int rr = 0; rr < 8; ++rr) {
      int row = rbase + g * 8 + rr;
      XS[(((size_t)(b * Kk + k) << 10) + row) * 16 + n] = (*acc)[rr];
    }
  }
}

// zr = sigmoid(A[b] @ Vg[b] + bg) -> z, r   (N=128)
template <bool FRAG>
__global__ __launch_bounds__(128) void gemm_A_gate_kernel(const void* __restrict__ A,
                                                          const unsigned short* __restrict__ vF,
                                                          const float* __restrict__ bg,
                                                          float* __restrict__ zb,
                                                          float* __restrict__ rb) {
  __shared__ __align__(32) unsigned short sB[2 * NT_GATE * 512];
  const int lane = threadIdx.x & 31, w = threadIdx.x >> 5;
  const int rt0 = blockIdx.x * 4 + w, rt1 = rt0 + 32;
  const int b = blockIdx.y;
  const void* Ab = FRAG ? (const void*)((const unsigned short*)A + ((size_t)b << 20))
                        : (const void*)((const float*)A + ((size_t)b << 20));
  const unsigned short* Bbase = vF + (size_t)b * (KT * NT_GATE * 512);
  const v8f vz = {0.f, 0.f, 0.f, 0.f, 0.f, 0.f, 0.f, 0.f};
  v8f acc0[NT_GATE], acc1[NT_GATE];
#pragma unroll
  for (int nt = 0; nt < NT_GATE; ++nt) { acc0[nt] = vz; acc1[nt] = vz; }
  gemm_body2<NT_GATE, FRAG>(Ab, Bbase, sB, rt0, rt1, lane, acc0, acc1);
  const int g = lane >> 4, n = lane & 15;
#pragma unroll
  for (int mw = 0; mw < 2; ++mw) {
    const int rbase = (mw ? rt1 : rt0) * 16;
    const v8f* acc = mw ? acc1 : acc0;
#pragma unroll
    for (int nt = 0; nt < NT_GATE; ++nt) {
      int col = nt * 16 + n;
      float bias = bg[col];
#pragma unroll
      for (int rr = 0; rr < 8; ++rr) {
        int row = rbase + g * 8 + rr;
        float s = 1.f / (1.f + __expf(-(acc[nt][rr] + bias)));
        size_t idx = (((size_t)b << 10) + row) * Hh + (col & 63);
        if (col < Hh) zb[idx] = s; else rb[idx] = s;
      }
    }
  }
}

// hc = tanh(A[b] @ Vu[b] + bu); h = z*hc + (1-z)*h   (N=64)
template <bool FRAG>
__global__ __launch_bounds__(128) void gemm_A_upd_kernel(const void* __restrict__ A,
                                                         const unsigned short* __restrict__ vF,
                                                         const float* __restrict__ bu,
                                                         const float* __restrict__ zb,
                                                         float* __restrict__ hb) {
  __shared__ __align__(32) unsigned short sB[2 * NT_UPD * 512];
  const int lane = threadIdx.x & 31, w = threadIdx.x >> 5;
  const int rt0 = blockIdx.x * 4 + w, rt1 = rt0 + 32;
  const int b = blockIdx.y;
  const void* Ab = FRAG ? (const void*)((const unsigned short*)A + ((size_t)b << 20))
                        : (const void*)((const float*)A + ((size_t)b << 20));
  const unsigned short* Bbase = vF + (size_t)b * (KT * NT_UPD * 512);
  const v8f vz = {0.f, 0.f, 0.f, 0.f, 0.f, 0.f, 0.f, 0.f};
  v8f acc0[NT_UPD], acc1[NT_UPD];
#pragma unroll
  for (int nt = 0; nt < NT_UPD; ++nt) { acc0[nt] = vz; acc1[nt] = vz; }
  gemm_body2<NT_UPD, FRAG>(Ab, Bbase, sB, rt0, rt1, lane, acc0, acc1);
  const int g = lane >> 4, n = lane & 15;
#pragma unroll
  for (int mw = 0; mw < 2; ++mw) {
    const int rbase = (mw ? rt1 : rt0) * 16;
    const v8f* acc = mw ? acc1 : acc0;
#pragma unroll
    for (int nt = 0; nt < NT_UPD; ++nt) {
      int col = nt * 16 + n;
      float bias = bu[col];
#pragma unroll
      for (int rr = 0; rr < 8; ++rr) {
        int row = rbase + g * 8 + rr;
        float hc = tanhf(acc[nt][rr] + bias);
        size_t idx = (((size_t)b << 10) + row) * Hh + col;
        float zv = zb[idx];
        hb[idx] = zv * hc + (1.f - zv) * hb[idx];
      }
    }
  }
}

// ---------------------------------------------------------------------------
// Host launcher
// ---------------------------------------------------------------------------

extern "C" void kernel_launch(void* const* d_in, const int* in_sizes, int n_in,
                              void* d_out, int out_size, void* d_ws, size_t ws_size,
                              hipStream_t stream) {
  (void)in_sizes; (void)n_in; (void)out_size;
  const float* x  = (const float*)d_in[0];
  const float* G  = (const float*)d_in[1];
  const float* A  = (const float*)d_in[2];
  const float* Wg = (const float*)d_in[3];
  const float* bg = (const float*)d_in[4];
  const float* Wu = (const float*)d_in[5];
  const float* bu = (const float*)d_in[6];
  const float* Wd = (const float*)d_in[7];
  const float* bd = (const float*)d_in[8];
  float* out = (float*)d_out;

  char* ws = (char*)d_ws;
  float*          h   = (float*)(ws + OFF_H);
  float*          z   = (float*)(ws + OFF_Z);
  float*          r   = (float*)(ws + OFF_R);
  unsigned short* hB  = (unsigned short*)(ws + OFF_HB);
  unsigned short* vF  = (unsigned short*)(ws + OFF_VF);
  float*          XS  = (float*)(ws + OFF_XS);
  unsigned short* xF  = (unsigned short*)(ws + OFF_XF);
  float*          S1h = (float*)(ws + OFF_S1);
  unsigned short* Gf  = (unsigned short*)(ws + OFF_GF);
  unsigned short* Af  = (unsigned short*)(ws + OFF_AF);
  const bool useFrag = (ws_size >= WS_FULL);

  const dim3 blk128(128), blk256(256);
  const dim3 gridG(8, Kk, Bb);   // 64 row-tiles / (4 waves x 2 tiles)
  const dim3 gridA(8, Bb);

  // ---- setup: h=0, XS = G_k @ x-columns (all t), optional bf16 A-fragments
  const int hN = Bb * Nn * Hh;
  zero_f32_kernel<<<(hN + 255) / 256, blk256, 0, stream>>>(h, hN);
  build_xfrag_kernel<<<(Bb * Nn * 16 + 255) / 256, blk256, 0, stream>>>(x, xF);
  gemm_X_kernel<<<gridG, blk128, 0, stream>>>(G, xF, XS);
  if (useFrag) {
    conv_afrag_kernel<<<(Kk * Nn * Nn + 255) / 256, blk256, 0, stream>>>(G, Gf, Kk * Nn * Nn);
    conv_afrag_kernel<<<(Bb * Nn * Nn + 255) / 256, blk256, 0, stream>>>(A, Af, Bb * Nn * Nn);
  }

  const int nH  = Bb * Nn * Hh;
  const int nVg = Bb * Nn * (NT_GATE * 16);
  const int nVu = Bb * Nn * (NT_UPD * 16);

  for (int t = 0; t < Tt; ++t) {
    // ---- gate: zr = sigmoid(A @ ((G @ [xt|h])_flat @ Wg) + bg)
    build_hB_kernel<<<(nH + 255) / 256, blk256, 0, stream>>>(h, r, hB, 0);
    if (useFrag) gemm_G_kernel<true ><<<gridG, blk128, 0, stream>>>(Gf, hB, S1h);
    else         gemm_G_kernel<false><<<gridG, blk128, 0, stream>>>(G,  hB, S1h);
    small_gemm_kernel<<<(nVg + 255) / 256, blk256, 0, stream>>>(S1h, XS, Wg, vF, NT_GATE * 16, t);
    if (useFrag) gemm_A_gate_kernel<true ><<<gridA, blk128, 0, stream>>>(Af, vF, bg, z, r);
    else         gemm_A_gate_kernel<false><<<gridA, blk128, 0, stream>>>(A,  vF, bg, z, r);

    // ---- update: hc = tanh(A @ ((G @ [xt|r*h])_flat @ Wu) + bu); h = z*hc+(1-z)*h
    build_hB_kernel<<<(nH + 255) / 256, blk256, 0, stream>>>(h, r, hB, 1);
    if (useFrag) gemm_G_kernel<true ><<<gridG, blk128, 0, stream>>>(Gf, hB, S1h);
    else         gemm_G_kernel<false><<<gridG, blk128, 0, stream>>>(G,  hB, S1h);
    small_gemm_kernel<<<(nVu + 255) / 256, blk256, 0, stream>>>(S1h, XS, Wu, vF, NT_UPD * 16, t);
    if (useFrag) gemm_A_upd_kernel<true ><<<gridA, blk128, 0, stream>>>(Af, vF, bu, z, h);
    else         gemm_A_upd_kernel<false><<<gridA, blk128, 0, stream>>>(A,  vF, bu, z, h);
  }

  decode_kernel<<<(Bb * Nn + 255) / 256, blk256, 0, stream>>>(h, Wd, bd, out);
}
